// Encoder1_25031069401690
// MI455X (gfx1250) — matrix operand order repaired and checked by reading, not compiled, not run on hardware
//
#include <hip/hip_runtime.h>

#define NN 50000
#define NE 800000
#define D 128
#define BN_EPS 1e-5f

typedef __attribute__((ext_vector_type(16))) _Float16 v16h;
typedef __attribute__((ext_vector_type(8)))  _Float16 v8h;
typedef __attribute__((ext_vector_type(8)))  float    v8f;

// ---------------- workspace layout (float units) ----------------
// [0      , 50000 )  deg_out
// [50000  , 100000)  deg_in
// [100000 , 100128)  bn_sum
// [100128 , 100256)  bn_sumsq
// [100256 , +6.4M )  agg  (later h2 in-place)
// [6500256, +6.4M )  h
// [12900256, +8192)  Bh   (f16 W fragments, 16384 halves)
#define ZERO_FLOATS 6500256  // deg_out..agg inclusive (divisible by 4)

__global__ void k_zero(float4* __restrict__ p, int n4) {
  int i = blockIdx.x * blockDim.x + threadIdx.x;
  if (i < n4) p[i] = make_float4(0.f, 0.f, 0.f, 0.f);
}

__global__ void k_deg(const int* __restrict__ src, const int* __restrict__ dst,
                      float* __restrict__ deg_out, float* __restrict__ deg_in) {
  int e = blockIdx.x * blockDim.x + threadIdx.x;
  if (e < NE) {
    atomicAdd(&deg_out[src[e]], 1.0f);
    atomicAdd(&deg_in[dst[e]],  1.0f);
  }
}

// Pre-swizzle W (f32 row-major [K=128][N=128]) into per-lane B fragments:
// fragment id f = (kc*8 + ntile), lane holds 16 consecutive halves.
// B fragment order used here (contiguous k, same as A): N = lane&15, half i -> K = kc*32 + (lane>>4)*16 + i.
__global__ void k_prepW(const float* __restrict__ W, _Float16* __restrict__ Bh) {
  int tid = blockIdx.x * blockDim.x + threadIdx.x;   // 0..1023
  int kc    = tid >> 8;
  int rem   = tid & 255;
  int ntile = rem >> 5;
  int lane  = rem & 31;
  int g     = lane >> 4;
  int n     = ntile * 16 + (lane & 15);
  _Float16* o = Bh + (size_t)tid * 16;
#pragma unroll
  for (int i = 0; i < 16; ++i) {
    int K = kc * 32 + g * 16 + i;
    o[i] = (_Float16)W[K * D + n];
  }
}

// h = (feat * rsqrt(max(out_deg,1))) @ W   via v_wmma_f32_16x16x32_f16
__global__ __launch_bounds__(256)
void k_gemm(const float* __restrict__ feat, const float* __restrict__ deg_out,
            const _Float16* __restrict__ Bh, float* __restrict__ h) {
  __shared__ _Float16 As[16 * 136];          // 136-half row pitch: conflict-free b128 reads
  const int row0 = blockIdx.x * 16;
  const int t = threadIdx.x;

  // stage 16x128 A tile (norm-scaled, f32->f16)
#pragma unroll
  for (int i = 0; i < 8; ++i) {
    int e = t + i * 256;
    int r = e >> 7, c = e & 127;
    float dg  = deg_out[row0 + r];
    float nrm = rsqrtf(dg < 1.0f ? 1.0f : dg);
    As[r * 136 + c] = (_Float16)(feat[(row0 + r) * D + c] * nrm);
  }
  __syncthreads();

  const int w    = t >> 5;        // N-tile index (0..7)
  const int lane = t & 31;
  const int M    = lane & 15;
  const int g    = lane >> 4;

  v8f acc = {};
#pragma unroll
  for (int kc = 0; kc < 4; ++kc) {
    // A fragment: half i -> K = kc*32 + g*16 + i
    v8h lo = *(const v8h*)&As[M * 136 + kc * 32 + g * 16];
    v8h hi = *(const v8h*)&As[M * 136 + kc * 32 + g * 16 + 8];
    v16h a = __builtin_shufflevector(lo, hi, 0,1,2,3,4,5,6,7,8,9,10,11,12,13,14,15);
    v16h b = *(const v16h*)(Bh + (size_t)((kc * 8 + w) * 32 + lane) * 16);
    acc = __builtin_amdgcn_wmma_f32_16x16x32_f16(false, a, false, b,
                                                 (short)0, acc, false, false);
  }

  // D layout: lane -> N = w*16 + (lane&15); VGPR r -> row M = r + 8*g
  const int ncol = w * 16 + M;
#pragma unroll
  for (int r = 0; r < 8; ++r) {
    h[(row0 + r + 8 * g) * D + ncol] = acc[r];
  }
}

// agg[dst] += h[src]  — one wave per edge, float4 per lane, L2-side atomics
__global__ __launch_bounds__(256)
void k_spmm(const int* __restrict__ src, const int* __restrict__ dst,
            const float* __restrict__ h, float* __restrict__ agg) {
  unsigned gt = blockIdx.x * 256u + threadIdx.x;
  int e    = (int)(gt >> 5);
  int lane = (int)(gt & 31);
  int s = src[e], d = dst[e];
  const float4 v = *(const float4*)(h + s * D + lane * 4);
  float* o = agg + d * D + lane * 4;
  atomicAdd(o + 0, v.x);
  atomicAdd(o + 1, v.y);
  atomicAdd(o + 2, v.z);
  atomicAdd(o + 3, v.w);
}

// h2 = prelu(agg*rsqrt(max(in_deg,1)) + b, a1) in-place; accumulate BN sums
__global__ __launch_bounds__(256)
void k_bnstat(float* __restrict__ agg, const float* __restrict__ deg_in,
              const float* __restrict__ bias, const float* __restrict__ a1,
              float* __restrict__ bn_sum, float* __restrict__ bn_sumsq) {
  __shared__ float s_sum[128], s_sq[128];
  const int t    = threadIdx.x;
  const int c    = t & 127;
  const int half = t >> 7;
  const float alpha = a1[0];
  const float bc    = bias[c];
  float sum = 0.f, sq = 0.f;
  const int n0 = blockIdx.x * 100 + half * 50;   // 500 blocks * 100 nodes = 50000
#pragma unroll 2
  for (int k = 0; k < 50; ++k) {
    int n = n0 + k;
    float dg  = deg_in[n];
    float nrm = rsqrtf(dg < 1.0f ? 1.0f : dg);
    float x = agg[n * D + c] * nrm + bc;
    x = (x >= 0.f) ? x : alpha * x;
    agg[n * D + c] = x;
    sum += x;
    sq  += x * x;
  }
  if (half) { s_sum[c] = sum; s_sq[c] = sq; }
  __syncthreads();
  if (!half) {
    atomicAdd(&bn_sum[c],   sum + s_sum[c]);
    atomicAdd(&bn_sumsq[c], sq  + s_sq[c]);
  }
}

__global__ __launch_bounds__(256)
void k_bnapply(const float* __restrict__ h2, const float* __restrict__ bn_sum,
               const float* __restrict__ bn_sumsq, const float* __restrict__ gamma,
               const float* __restrict__ beta, const float* __restrict__ a2,
               float* __restrict__ out) {
  int i = blockIdx.x * blockDim.x + threadIdx.x;
  if (i >= NN * D) return;
  int c = i & 127;
  const float invN = 1.0f / (float)NN;
  float mean = bn_sum[c] * invN;
  float var  = bn_sumsq[c] * invN - mean * mean;
  var = (var < 0.f) ? 0.f : var;
  float inv = rsqrtf(var + BN_EPS);
  float y = (h2[i] - mean) * inv * gamma[c] + beta[c];
  float alpha = a2[0];
  out[i] = (y >= 0.f) ? y : alpha * y;
}

extern "C" void kernel_launch(void* const* d_in, const int* in_sizes, int n_in,
                              void* d_out, int out_size, void* d_ws, size_t ws_size,
                              hipStream_t stream) {
  const float* feat  = (const float*)d_in[0];
  const int*   src   = (const int*)d_in[1];
  const int*   dst   = (const int*)d_in[2];
  const float* W     = (const float*)d_in[3];
  const float* bias  = (const float*)d_in[4];
  const float* a1    = (const float*)d_in[5];
  const float* gamma = (const float*)d_in[6];
  const float* beta  = (const float*)d_in[7];
  const float* a2    = (const float*)d_in[8];
  float* out = (float*)d_out;

  float* ws       = (float*)d_ws;
  float* deg_out  = ws;
  float* deg_in   = ws + 50000;
  float* bn_sum   = ws + 100000;
  float* bn_sumsq = ws + 100128;
  float* agg      = ws + 100256;
  float* h        = ws + 100256 + (size_t)NN * D;
  _Float16* Bh    = (_Float16*)(ws + 100256 + 2 * (size_t)NN * D);

  // zero accumulators (deg_out, deg_in, bn sums, agg)
  const int n4 = ZERO_FLOATS / 4;
  k_zero<<<(n4 + 255) / 256, 256, 0, stream>>>((float4*)ws, n4);
  k_deg<<<(NE + 255) / 256, 256, 0, stream>>>(src, dst, deg_out, deg_in);
  k_prepW<<<4, 256, 0, stream>>>(W, Bh);
  k_gemm<<<NN / 16, 256, 0, stream>>>(feat, deg_out, Bh, h);   // 3125 full tiles
  k_spmm<<<(NE * 32) / 256, 256, 0, stream>>>(src, dst, h, agg); // 100000 blocks
  k_bnstat<<<500, 256, 0, stream>>>(agg, deg_in, bias, a1, bn_sum, bn_sumsq);
  k_bnapply<<<(NN * D + 255) / 256, 256, 0, stream>>>(agg, bn_sum, bn_sumsq,
                                                      gamma, beta, a2, out);
}
